// MACE_41987600285860
// MI455X (gfx1250) — matrix operand order repaired and checked by reading, not compiled, not run on hardware
//
#include <hip/hip_runtime.h>
#include <hip/hip_bf16.h>

typedef _Float16 f16;
typedef __attribute__((ext_vector_type(16))) _Float16 v16h;
typedef __attribute__((ext_vector_type(8)))  _Float16 v8h;
typedef __attribute__((ext_vector_type(8)))  float    v8f;

#define NNODE 10000
#define NEDGE 160000
#define NCH 64

static __device__ __forceinline__ v8f wmma16(v16h a, v16h b, v8f c) {
  return __builtin_amdgcn_wmma_f32_16x16x32_f16(false, a, false, b, (short)0, c, false, false);
}

// Build 16-bit A-fragment (16 rows x 32 K), contiguous K order:
//   lanes 0-15 : row=lane,    elems 0-15 = K kbase+0..15
//   lanes 16-31: row=lane-16, elems 0-15 = K kbase+16..31
static __device__ __forceinline__ v16h make_a(const f16* rowp, int kbase, int koff) {
  v8h lo = *(const v8h*)(rowp + kbase + 2 * koff);
  v8h hi = *(const v8h*)(rowp + kbase + 2 * koff + 8);
  v16h a;
#pragma unroll
  for (int i = 0; i < 8; ++i) { a[i] = lo[i]; a[8 + i] = hi[i]; }
  return a;
}

// fast SiLU: x * sigmoid(x) with hardware v_rcp_f32 (avoids div_scale chain)
static __device__ __forceinline__ float silu(float x) {
  return x * __builtin_amdgcn_rcpf(1.0f + __expf(-x));
}

// ---------------- zero workspace + output ----------------
__global__ void zero_kernel(float* __restrict__ agg, float* __restrict__ out) {
  int idx = blockIdx.x * 256 + threadIdx.x;
  if (idx < NNODE * 576) agg[idx] = 0.0f;
  if (idx < 64) out[idx] = 0.0f;
}

// ---------------- edge geometry: sh[E,9] f32, rad[E,8] f16 ----------------
__global__ void geom_kernel(const float* __restrict__ pos,
                            const int* __restrict__ senders,
                            const int* __restrict__ receivers,
                            float* __restrict__ sh, f16* __restrict__ radh) {
  int e = blockIdx.x * 256 + threadIdx.x;
  if (e >= NEDGE) return;
  int s = senders[e], rc = receivers[e];
  float vx = pos[rc * 3 + 0] - pos[s * 3 + 0];
  float vy = pos[rc * 3 + 1] - pos[s * 3 + 1];
  float vz = pos[rc * 3 + 2] - pos[s * 3 + 2];
  float r = sqrtf(vx * vx + vy * vy + vz * vz);
  float rs = fmaxf(r, 1e-9f);
  float inv = 1.0f / rs;
  float x = vx * inv, y = vy * inv, z = vz * inv;
  const float s3 = 1.7320508075688772f;
  const float s15 = 3.872983346207417f;
  const float s5 = 2.23606797749979f;
  float* o = sh + (size_t)e * 9;
  o[0] = 1.0f;
  o[1] = s3 * x; o[2] = s3 * y; o[3] = s3 * z;
  o[4] = s15 * x * y; o[5] = s15 * y * z;
  o[6] = 0.5f * s5 * (3.0f * z * z - 1.0f);
  o[7] = s15 * x * z;
  o[8] = 0.5f * s15 * (x * x - y * y);
  // Bessel (n=1..8) * polynomial cutoff p=5, R=5
  float xr = r * 0.2f;
  float xr2 = xr * xr, xr4 = xr2 * xr2, xr5 = xr4 * xr;
  float env = (xr < 1.0f)
      ? (1.0f - 21.0f * xr5 + 35.0f * xr5 * xr - 15.0f * xr5 * xr2) : 0.0f;
  float pref = 0.6324555320336759f * inv * env;  // sqrt(2/5)/r_safe*env
  const float w = 3.14159265358979f * 0.2f;      // pi/R
  f16* rh = radh + (size_t)e * 8;
#pragma unroll
  for (int n = 1; n <= 8; ++n)
    rh[n - 1] = (f16)(pref * __sinf((float)n * w * rs));
}

// ---------------- node features xg[n,ch] = emb[species[n],ch] -------------
__global__ void feat_kernel(const float* __restrict__ emb,
                            const int* __restrict__ species,
                            float* __restrict__ xg) {
  int idx = blockIdx.x * 256 + threadIdx.x;
  if (idx >= NNODE * NCH) return;
  int n = idx >> 6, c = idx & 63;
  xg[idx] = emb[species[n] * NCH + c];
}

// ---- fused edge MLP (WMMA, f16 in / f32 acc) + message + scatter-add ----
__global__ __launch_bounds__(256) void mlp_msg_kernel(
    const f16* __restrict__ radh, const float* __restrict__ sh,
    const float* __restrict__ xg,
    const float* __restrict__ w1, const float* __restrict__ w2,
    const float* __restrict__ w3, const float* __restrict__ w4,
    const int* __restrict__ senders, const int* __restrict__ receivers,
    float* __restrict__ agg) {
  __shared__ alignas(128) f16 w1T[64][32];      // [out][k], k>=8 zero
  __shared__ alignas(128) f16 w2T[64][64];      // [out][k]
  __shared__ alignas(128) f16 w3T[64][64];
  __shared__ alignas(128) f16 w4T[192][64];
  __shared__ alignas(128) f16 stage[8][16][64]; // per-wave 16-row tile

  const int tid = threadIdx.x;
  for (int idx = tid; idx < 64 * 32; idx += 256) {
    int c = idx >> 5, k = idx & 31;
    w1T[c][k] = (k < 8) ? (f16)w1[k * 64 + c] : (f16)0.0f;
  }
  for (int idx = tid; idx < 64 * 64; idx += 256) {
    int c = idx >> 6, k = idx & 63;
    w2T[c][k] = (f16)w2[k * 64 + c];
    w3T[c][k] = (f16)w3[k * 64 + c];
  }
  for (int idx = tid; idx < 192 * 64; idx += 256) {
    int c = idx >> 6, k = idx & 63;
    w4T[c][k] = (f16)w4[k * 192 + c];
  }

  const int wv = tid >> 5, lane = tid & 31;
  const int laneHi = lane >> 4, ln = lane & 15;
  const int koff = laneHi ? 8 : 0, kselB = laneHi ? 16 : 0;
  const int eb = blockIdx.x * 128 + wv * 16;   // NEDGE % 128 == 0: always full
  f16(*st)[64] = stage[wv];

  if (lane < 16) {                 // stage radial features, zero-pad K to 32
    int e = eb + lane;
    *(v8h*)(&st[lane][0]) = *(const v8h*)(radh + (size_t)e * 8);
    v8h z = {};
    *(v8h*)(&st[lane][8]) = z;
    *(v8h*)(&st[lane][16]) = z;
    *(v8h*)(&st[lane][24]) = z;
  }
  __syncthreads();

  // ---- layer 1: K=8 (padded to 32), N=64 ----
  {
    v16h a0 = make_a(&st[ln][0], 0, koff);
    float outv[4][8];
#pragma unroll
    for (int t = 0; t < 4; ++t) {
      v16h b = *(const v16h*)(&w1T[t * 16 + ln][kselB]);
      v8f acc = {};
      acc = wmma16(a0, b, acc);
#pragma unroll
      for (int i = 0; i < 8; ++i) outv[t][i] = silu(acc[i]);
    }
    __syncthreads();
#pragma unroll
    for (int t = 0; t < 4; ++t)
#pragma unroll
      for (int i = 0; i < 8; ++i)
        st[i + laneHi * 8][t * 16 + ln] = (f16)outv[t][i];
    __syncthreads();
  }

  // ---- layers 2,3: K=64, N=64 ----
#pragma unroll 1
  for (int layer = 0; layer < 2; ++layer) {
    f16(*wT)[64] = (layer == 0) ? w2T : w3T;
    v16h a0 = make_a(&st[ln][0], 0, koff);
    v16h a1 = make_a(&st[ln][0], 32, koff);
    float outv[4][8];
#pragma unroll
    for (int t = 0; t < 4; ++t) {
      v16h b0 = *(const v16h*)(&wT[t * 16 + ln][kselB]);
      v16h b1 = *(const v16h*)(&wT[t * 16 + ln][32 + kselB]);
      v8f acc = {};
      acc = wmma16(a0, b0, acc);
      acc = wmma16(a1, b1, acc);
#pragma unroll
      for (int i = 0; i < 8; ++i) outv[t][i] = silu(acc[i]);
    }
    __syncthreads();
#pragma unroll
    for (int t = 0; t < 4; ++t)
#pragma unroll
      for (int i = 0; i < 8; ++i)
        st[i + laneHi * 8][t * 16 + ln] = (f16)outv[t][i];
    __syncthreads();
  }

  // ---- layer 4 (K=64, N=192) fused with message scatter ----
  {
    v16h a0 = make_a(&st[ln][0], 0, koff);
    v16h a1 = make_a(&st[ln][0], 32, koff);

    // hoist per-edge metadata (same 8 edge rows for every output tile)
    int se[8], rcv[8];
#pragma unroll
    for (int i = 0; i < 8; ++i) {
      int e = eb + i + laneHi * 8;
      se[i] = senders[e];
      rcv[i] = receivers[e];
    }

#pragma unroll 1
    for (int t = 0; t < 12; ++t) {
      v16h b0 = *(const v16h*)(&w4T[t * 16 + ln][kselB]);
      v16h b1 = *(const v16h*)(&w4T[t * 16 + ln][32 + kselB]);
      v8f acc = {};
      acc = wmma16(a0, b0, acc);
      acc = wmma16(a1, b1, acc);
      int j = t * 16 + ln;          // column of h@w4 -> (ch, l) = (j/3, j%3)
      int ch = j / 3, l = j - ch * 3;
#pragma unroll
      for (int i = 0; i < 8; ++i) {
        int e = eb + i + laneHi * 8;
        float coef = xg[(size_t)se[i] * NCH + ch] * acc[i];
        const float* shp = sh + (size_t)e * 9;
        float* ap = agg + (size_t)rcv[i] * 576 + (size_t)ch * 9;
        if (l == 0) {
          atomicAdd(ap + 0, coef * shp[0]);
        } else if (l == 1) {
          atomicAdd(ap + 1, coef * shp[1]);
          atomicAdd(ap + 2, coef * shp[2]);
          atomicAdd(ap + 3, coef * shp[3]);
        } else {
#pragma unroll
          for (int m = 4; m < 9; ++m) atomicAdd(ap + m, coef * shp[m]);
        }
      }
    }
  }
}

// ---- node mixing (WMMA), invariant B0, readout, graph segment-sum ----
__global__ __launch_bounds__(256) void node_kernel(
    const float* __restrict__ agg, const float* __restrict__ mix_w,
    const float* __restrict__ prod_w, const float* __restrict__ readout_w,
    const float* __restrict__ atomic_energies,
    const int* __restrict__ species, const int* __restrict__ batch_ids,
    float* __restrict__ out) {
  __shared__ alignas(128) f16 mwT[3][64][64];   // [l][c][k] = mix_w[l][k][c]
  __shared__ alignas(128) f16 stage[8][16][64];

  const int tid = threadIdx.x;
  for (int idx = tid; idx < 3 * 64 * 64; idx += 256) {
    int l = idx >> 12, r = idx & 4095, c = r >> 6, k = r & 63;
    mwT[l][c][k] = (f16)mix_w[l * 4096 + k * 64 + c];
  }
  __syncthreads();

  const int wv = tid >> 5, lane = tid & 31;
  const int laneHi = lane >> 4, ln = lane & 15;
  const int koff = laneHi ? 8 : 0, kselB = laneHi ? 16 : 0;
  const int nb = blockIdx.x * 128 + wv * 16;
  f16(*st)[64] = stage[wv];

  float eacc[4][8];
#pragma unroll
  for (int t = 0; t < 4; ++t)
#pragma unroll
    for (int i = 0; i < 8; ++i) eacc[t][i] = 0.0f;

  const int LofM[9] = {0, 1, 1, 1, 2, 2, 2, 2, 2};
#pragma unroll 1
  for (int m = 0; m < 9; ++m) {
    int l = LofM[m];
    for (int idx = lane; idx < 1024; idx += 32) {   // stage agg(:,:,m)/16
      int r = idx >> 6, k = idx & 63;
      int n = nb + r; if (n >= NNODE) n = NNODE - 1;
      st[r][k] = (f16)(agg[(size_t)n * 576 + (size_t)k * 9 + m] * 0.0625f);
    }
    __syncthreads();
    v16h a0 = make_a(&st[ln][0], 0, koff);
    v16h a1 = make_a(&st[ln][0], 32, koff);
#pragma unroll
    for (int t = 0; t < 4; ++t) {
      v16h b0 = *(const v16h*)(&mwT[l][t * 16 + ln][kselB]);
      v16h b1 = *(const v16h*)(&mwT[l][t * 16 + ln][32 + kselB]);
      v8f acc = {};
      acc = wmma16(a0, b0, acc);
      acc = wmma16(a1, b1, acc);
      int ch = t * 16 + ln;
      float pwv = prod_w[l * 64 + ch];
#pragma unroll
      for (int i = 0; i < 8; ++i) {
        float A = acc[i];
        float contrib = A * A * pwv;       // B0 term
        if (m == 0) contrib += A;          // scalar channel A[:,:,0]
        eacc[t][i] += contrib;
      }
    }
    __syncthreads();
  }

  float rw[4];
#pragma unroll
  for (int t = 0; t < 4; ++t) rw[t] = readout_w[t * 16 + ln];
#pragma unroll
  for (int i = 0; i < 8; ++i) {
    float s = eacc[0][i] * rw[0] + eacc[1][i] * rw[1] +
              eacc[2][i] * rw[2] + eacc[3][i] * rw[3];
    s += __shfl_xor(s, 1);
    s += __shfl_xor(s, 2);
    s += __shfl_xor(s, 4);
    s += __shfl_xor(s, 8);     // sums 16 channel-lanes within each half
    if (ln == 0) {
      int n = nb + i + laneHi * 8;
      if (n < NNODE) {
        float val = s + atomic_energies[species[n]];
        atomicAdd(out + batch_ids[n], val);
      }
    }
  }
}

extern "C" void kernel_launch(void* const* d_in, const int* in_sizes, int n_in,
                              void* d_out, int out_size, void* d_ws, size_t ws_size,
                              hipStream_t stream) {
  const float* pos      = (const float*)d_in[0];
  const int*   species  = (const int*)d_in[1];
  const int*   senders  = (const int*)d_in[2];
  const int*   receivers= (const int*)d_in[3];
  const int*   batch    = (const int*)d_in[4];
  const float* emb      = (const float*)d_in[5];
  const float* w1       = (const float*)d_in[6];
  const float* w2       = (const float*)d_in[7];
  const float* w3       = (const float*)d_in[8];
  const float* w4       = (const float*)d_in[9];
  const float* mix_w    = (const float*)d_in[10];
  const float* prod_w   = (const float*)d_in[11];
  const float* readout  = (const float*)d_in[12];
  const float* ae       = (const float*)d_in[13];
  float* out = (float*)d_out;

  char* base = (char*)d_ws;
  float* shb  = (float*)(base);                     // E*9*4  = 5,760,000 B
  f16*   radh = (f16*)(base + 5760000);             // E*8*2  = 2,560,000 B
  float* xg   = (float*)(base + 8320000);           // N*64*4 = 2,560,000 B
  float* agg  = (float*)(base + 10880000);          // N*576*4= 23,040,000 B

  zero_kernel<<<(NNODE * 576 + 255) / 256, 256, 0, stream>>>(agg, out);
  geom_kernel<<<(NEDGE + 255) / 256, 256, 0, stream>>>(pos, senders, receivers, shb, radh);
  feat_kernel<<<(NNODE * NCH + 255) / 256, 256, 0, stream>>>(emb, species, xg);
  mlp_msg_kernel<<<NEDGE / 128, 256, 0, stream>>>(radh, shb, xg, w1, w2, w3, w4,
                                                  senders, receivers, agg);
  node_kernel<<<(NNODE + 127) / 128, 256, 0, stream>>>(agg, mix_w, prod_w, readout,
                                                       ae, species, batch, out);
}
